// Decoder_36867999269325
// MI455X (gfx1250) — matrix operand order repaired and checked by reading, not compiled, not run on hardware
//
#include <hip/hip_runtime.h>
#include <hip/hip_fp16.h>

typedef __attribute__((ext_vector_type(16))) _Float16 v16h;
typedef __attribute__((ext_vector_type(8)))  float    v8f;
typedef __attribute__((ext_vector_type(4)))  unsigned int u32x4;
typedef __attribute__((ext_vector_type(8)))  int      i32x8;
typedef __attribute__((ext_vector_type(4)))  int      i32x4;

#define WMMA_F16(a, b, c) \
  __builtin_amdgcn_wmma_f32_16x16x32_f16(false, (a), false, (b), (short)0, (c), false, false)

constexpr int N  = 10000;
constexpr int T  = 12;
constexpr int NF = 16;
constexpr int H  = 128;
constexpr int L  = 64;
constexpr int E  = 160000;
constexpr int TN = T * N;   // 120000, multiple of 16

#if __has_builtin(__builtin_amdgcn_tensor_load_to_lds) && \
    __has_builtin(__builtin_amdgcn_s_wait_tensorcnt)
#define USE_TDM 1
#else
#define USE_TDM 0
#endif

// ---------------- Tensor Data Mover: 2D f16 tile (tile == tensor) -> LDS ----------------
// D# per CDNA5 ISA: group0 = {count, lds_addr, global_addr, type=2},
// group1 = {data_size=2B, tensor_dim0/1, tile_dim0/1, strides}. Issued by one wave per
// block (EXEC ignored by TDM); helper waits TENSORcnt, caller then barriers.
#if USE_TDM
__device__ inline void tdm_load_2d_f16(const _Float16* src, unsigned lds_off,
                                       unsigned dim0, unsigned dim1) {
  unsigned long long ga = (unsigned long long)(uintptr_t)src;
  u32x4 g0;
  g0[0] = 1u;                                            // count=1, user mode
  g0[1] = lds_off;                                       // LDS byte address
  g0[2] = (unsigned)ga;                                  // global_addr[31:0]
  g0[3] = (unsigned)((ga >> 32) & 0x1FFFFFFu) | (2u << 30);  // addr[56:32] | type=2
  unsigned long long s1 = (unsigned long long)dim0 * dim1;   // dim1 stride (unused, 2D)
  i32x8 g1;
  g1[0] = (int)(1u << 16);                               // data_size code 1 = 2 bytes
  g1[1] = (int)((dim0 & 0xFFFFu) << 16);                 // tensor_dim0[15:0]
  g1[2] = (int)(((dim0 >> 16) & 0xFFFFu) | ((dim1 & 0xFFFFu) << 16));
  g1[3] = (int)(((dim1 >> 16) & 0xFFFFu) | ((dim0 & 0xFFFFu) << 16)); // tile_dim0 = dim0
  g1[4] = (int)(dim1 & 0xFFFFu);                         // tile_dim1 = dim1, tile_dim2=0
  g1[5] = (int)dim0;                                     // tensor_dim0_stride[31:0]
  g1[6] = (int)((s1 & 0xFFFFu) << 16);                   // stride0[47:32]=0 | stride1[15:0]
  g1[7] = (int)((s1 >> 16) & 0xFFFFFFFFu);               // stride1[47:16]
  i32x4 g2 = {0, 0, 0, 0};
  i32x4 g3 = {0, 0, 0, 0};
#if __has_include(<hip/amd_detail/amd_gfx1250_TDM.h>)
  i32x8 g4 = {0, 0, 0, 0, 0, 0, 0, 0};
  __builtin_amdgcn_tensor_load_to_lds(g0, g1, g2, g3, g4, 0);
#else
  __builtin_amdgcn_tensor_load_to_lds(g0, g1, g2, g3, 0);
#endif
  __builtin_amdgcn_s_wait_tensorcnt(0);
}
#endif

// ---------------- WMMA fragment loaders (wave32, 16x16x32 f16) ----------------

// A fragment: 16x32 f16 tile; per-lane row pointer already at K=0 of this tile.
__device__ inline v16h load_a_frag_row(const _Float16* __restrict__ rowk0) {
  int lane = threadIdx.x & 31;
  int kb = (lane >> 4) << 3;           // lanes 16-31 hold K offset +8
  v16h a;
#pragma unroll
  for (int v = 0; v < 8; ++v) {
    int k = ((v < 4) ? (2 * v) : (16 + 2 * (v - 4))) + kb;
    a[2 * v]     = rowk0[k];
    a[2 * v + 1] = rowk0[k + 1];
  }
  return a;
}

__device__ inline v16h load_a_frag(const _Float16* __restrict__ A, int lda) {
  int m = threadIdx.x & 15;
  return load_a_frag_row(A + (size_t)m * lda);
}

// B fragment: 32x16 f16 tile; lane%16 = column N; elements over K in table order.
__device__ inline v16h load_b_frag(const _Float16* __restrict__ B, int ldb) {
  int lane = threadIdx.x & 31;
  int n = lane & 15, kb = (lane >> 4) << 3;
  v16h b;
#pragma unroll
  for (int j = 0; j < 16; ++j) b[j] = B[(size_t)(((j < 8) ? j : j + 8) + kb) * ldb + n];
  return b;
}

// ---------------- small prep kernels ----------------

__global__ void k_conv_weights(const float* __restrict__ Wfc2, const float* __restrict__ Wih,
                               const float* __restrict__ Whh,  const float* __restrict__ Wgcn,
                               const float* __restrict__ Wfc3,
                               _Float16* Wfc2h, _Float16* WihTh, _Float16* WhhTh,
                               _Float16* Wgcnh, _Float16* Wfc3h) {
  int i = blockIdx.x * blockDim.x + threadIdx.x;      // launched with 65536 threads
  if (i < L * H)  Wfc2h[i] = (_Float16)Wfc2[i];
  if (i < 4 * H * H) {
    int j = i / H, k = i % H;                          // W[j][k] -> B[k][j]
    WihTh[k * (4 * H) + j] = (_Float16)Wih[i];
    WhhTh[k * (4 * H) + j] = (_Float16)Whh[i];
  }
  if (i < H * H)  Wgcnh[i] = (_Float16)Wgcn[i];
  if (i < H * NF) Wfc3h[i] = (_Float16)Wfc3[i];
}

__global__ void k_f2h(const float* __restrict__ s, _Float16* __restrict__ d, long total) {
  for (long i = (long)blockIdx.x * blockDim.x + threadIdx.x; i < total;
       i += (long)gridDim.x * blockDim.x)
    d[i] = (_Float16)s[i];
}

__global__ void k_nodemask(const unsigned char* __restrict__ xm, float* __restrict__ nmask) {
  int n = blockIdx.x * blockDim.x + threadIdx.x;
  if (n >= N) return;
  const unsigned int* p = (const unsigned int*)(xm + (size_t)n * (T * NF));
  unsigned int any = 0;
#pragma unroll 4
  for (int i = 0; i < (T * NF) / 4; ++i) any |= p[i];
  nmask[n] = any ? 1.0f : 0.0f;
}

__global__ void k_deg_init(float* deg) {
  int n = blockIdx.x * blockDim.x + threadIdx.x;
  if (n < N) deg[n] = 1.0f;                            // self-loop
}
__global__ void k_deg_acc(const int* __restrict__ ei, float* deg) {
  int e = blockIdx.x * blockDim.x + threadIdx.x;
  if (e < E) atomicAdd(&deg[ei[E + e]], 1.0f);         // dst row
}
__global__ void k_deg_fin(float* deg) {
  int n = blockIdx.x * blockDim.x + threadIdx.x;
  if (n < N) deg[n] = rsqrtf(deg[n]);
}

// ---------------- GEMM kernels ----------------

// hd = z @ W_fc2 + b_fc2 ; also h0 := hd (f16), c := 0
__global__ void k_gemm_fc2(const _Float16* __restrict__ zh, const _Float16* __restrict__ Wb,
                           const float* __restrict__ bias, _Float16* hdh, _Float16* h0,
                           float* c) {
  int wave = threadIdx.x >> 5, lane = threadIdx.x & 31;
  int rt = blockIdx.x, ct = wave;                      // 625 x 8 tiles
  int n = lane & 15, hi = lane >> 4;
  float bv = bias[ct * 16 + n];
  v8f acc;
#pragma unroll
  for (int r = 0; r < 8; ++r) acc[r] = bv;
#pragma unroll
  for (int kk = 0; kk < 2; ++kk) {                     // K = 64
    v16h a = load_a_frag(zh + (size_t)rt * 16 * L + kk * 32, L);
    v16h b = load_b_frag(Wb + (size_t)kk * 32 * H + ct * 16, H);
    acc = WMMA_F16(a, b, acc);
  }
#pragma unroll
  for (int r = 0; r < 8; ++r) {
    size_t off = (size_t)(rt * 16 + hi * 8 + r) * H + ct * 16 + n;
    _Float16 hv = (_Float16)acc[r];
    hdh[off] = hv; h0[off] = hv; c[off] = 0.0f;
  }
}

// x_proj = hd @ W_ih^T + b_ih   [N, 512]
__global__ void k_gemm_xproj(const _Float16* __restrict__ hdh, const _Float16* __restrict__ WihTh,
                             const float* __restrict__ bih, float* __restrict__ xproj) {
  int wave = threadIdx.x >> 5, lane = threadIdx.x & 31;
  int rt = blockIdx.x;
  int ct = blockIdx.y * 8 + wave;                      // 0..31
  int n = lane & 15, hi = lane >> 4;
  float bv = bih[ct * 16 + n];
  v8f acc;
#pragma unroll
  for (int r = 0; r < 8; ++r) acc[r] = bv;
#pragma unroll
  for (int kk = 0; kk < 4; ++kk) {                     // K = 128
    v16h a = load_a_frag(hdh + (size_t)rt * 16 * H + kk * 32, H);
    v16h b = load_b_frag(WihTh + (size_t)kk * 32 * (4 * H) + ct * 16, 4 * H);
    acc = WMMA_F16(a, b, acc);
  }
#pragma unroll
  for (int r = 0; r < 8; ++r)
    xproj[(size_t)(rt * 16 + hi * 8 + r) * (4 * H) + ct * 16 + n] = acc[r];
}

// One LSTM step: gates = x_proj + h @ W_hh^T + b_hh; update c,h; emit masked xt (f16).
// W_hh^T (128 KB f16) is staged in LDS once per block via the Tensor Data Mover.
__global__ void k_lstm_step(const _Float16* __restrict__ hin, const _Float16* __restrict__ WhhTh,
                            const float* __restrict__ xproj, const float* __restrict__ bhh,
                            float* __restrict__ c, _Float16* __restrict__ hout,
                            const float* __restrict__ nmask, _Float16* __restrict__ xth, int t) {
  extern __shared__ _Float16 sW[];                     // 128 x 512 f16 = 128 KB
  int wave = threadIdx.x >> 5, lane = threadIdx.x & 31;
  int rt = blockIdx.x, ct = wave;                      // 625 x 8 (H col tiles)
  int n = lane & 15, hi = lane >> 4;

#if USE_TDM
  if (threadIdx.x == 0)
    tdm_load_2d_f16(WhhTh, (unsigned)(uintptr_t)sW, 4 * H, H);
#else
  for (int i = threadIdx.x; i < H * 4 * H / 8; i += blockDim.x)
    ((float4*)sW)[i] = ((const float4*)WhhTh)[i];
#endif
  __syncthreads();

  v16h a[4];
#pragma unroll
  for (int kk = 0; kk < 4; ++kk)
    a[kk] = load_a_frag(hin + (size_t)rt * 16 * H + kk * 32, H);

  v8f g[4];
#pragma unroll
  for (int gi = 0; gi < 4; ++gi) {                     // i, f, g, o
    int col = gi * H + ct * 16 + n;
    float bv = bhh[col];
    v8f acc;
#pragma unroll
    for (int r = 0; r < 8; ++r)
      acc[r] = xproj[(size_t)(rt * 16 + hi * 8 + r) * (4 * H) + col] + bv;
#pragma unroll
    for (int kk = 0; kk < 4; ++kk) {
      v16h b;                                          // B frag from LDS (ds_load)
#pragma unroll
      for (int j = 0; j < 16; ++j)
        b[j] = sW[(size_t)(kk * 32 + ((j < 8) ? j : j + 8) + hi * 8) * (4 * H) + gi * H + ct * 16 + n];
      acc = WMMA_F16(a[kk], b, acc);
    }
    g[gi] = acc;
  }

#pragma unroll
  for (int r = 0; r < 8; ++r) {
    int row = rt * 16 + hi * 8 + r;
    size_t off = (size_t)row * H + ct * 16 + n;
    float iv = 1.0f / (1.0f + __expf(-g[0][r]));
    float fv = 1.0f / (1.0f + __expf(-g[1][r]));
    float gv = tanhf(g[2][r]);
    float ov = 1.0f / (1.0f + __expf(-g[3][r]));
    float cn = fv * c[off] + iv * gv;
    float hn = ov * tanhf(cn);
    c[off] = cn;
    hout[off] = (_Float16)hn;
    float o = hn * nmask[row];
    if (isnan(o)) o = 0.0f;
    xth[(size_t)t * N * H + off] = (_Float16)o;
  }
}

// xw = xt @ W_gcn over flattened [T*N, H]; epilogue fused: agg = b_gcn + dinv[n]^2 * xw
__global__ void k_gemm_gcn(const _Float16* __restrict__ xth, const _Float16* __restrict__ Wg,
                           const float* __restrict__ dinv, const float* __restrict__ bgcn,
                           float* __restrict__ xw, float* __restrict__ agg) {
  __shared__ _Float16 sB[H * H];                       // 32 KB
  int wave = threadIdx.x >> 5, lane = threadIdx.x & 31;
  int rt = blockIdx.x, ct = wave;                      // 7500 x 8
  int n = lane & 15, hi = lane >> 4;

#if USE_TDM
  if (threadIdx.x == 0)
    tdm_load_2d_f16(Wg, (unsigned)(uintptr_t)sB, H, H);
#else
  for (int i = threadIdx.x; i < H * H / 8; i += blockDim.x)
    ((float4*)sB)[i] = ((const float4*)Wg)[i];
#endif
  __syncthreads();

  v8f acc;
#pragma unroll
  for (int r = 0; r < 8; ++r) acc[r] = 0.0f;
#pragma unroll
  for (int kk = 0; kk < 4; ++kk) {
    v16h a = load_a_frag(xth + (size_t)rt * 16 * H + kk * 32, H);
    v16h b;                                            // B frag from LDS (ds_load)
#pragma unroll
    for (int j = 0; j < 16; ++j)
      b[j] = sB[(size_t)(kk * 32 + ((j < 8) ? j : j + 8) + hi * 8) * H + ct * 16 + n];
    acc = WMMA_F16(a, b, acc);
  }
  float bg = bgcn[ct * 16 + n];
#pragma unroll
  for (int r = 0; r < 8; ++r) {
    int row = rt * 16 + hi * 8 + r;                    // flat t*N + n
    int nn = row % N;
    float d = dinv[nn];
    size_t off = (size_t)row * H + ct * 16 + n;
    float v = acc[r];
    xw[off] = v;
    agg[off] = bg + d * d * v;                         // self-loop + bias folded in
  }
}

// agg[t, dst, :] += dinv[src]*dinv[dst] * xw[t, src, :]  (atomic, float4 per thread)
__global__ void k_scatter(const int* __restrict__ ei, const float* __restrict__ dinv,
                          const float* __restrict__ xw, float* __restrict__ agg) {
  long tid = (long)blockIdx.x * blockDim.x + threadIdx.x;
  long total = (long)E * T * (H / 4);
  if (tid >= total) return;
  int q = (int)(tid & 31);                 // 32 quads cover H=128
  long r = tid >> 5;
  int e = (int)(r % E);
  int t = (int)(r / E);
  int s = ei[e], d = ei[E + e];
  float nm = dinv[s] * dinv[d];
  const float4 v = *(const float4*)(xw + ((size_t)t * N + s) * H + q * 4);
  float* o = agg + ((size_t)t * N + d) * H + q * 4;
  atomicAdd(o + 0, nm * v.x);
  atomicAdd(o + 1, nm * v.y);
  atomicAdd(o + 2, nm * v.z);
  atomicAdd(o + 3, nm * v.w);
}

// x_hat[n, t, :] = spatial[t, n, :] @ W_fc3 + b_fc3   (rows flattened as n*T+t)
__global__ void k_gemm_fc3(const _Float16* __restrict__ spath, const _Float16* __restrict__ Wf,
                           const float* __restrict__ bfc3, float* __restrict__ out) {
  int wave = threadIdx.x >> 5, lane = threadIdx.x & 31;
  int rt = blockIdx.x * 8 + wave;                      // 7500 row tiles, 1 col tile
  if (rt >= TN / 16) return;                           // uniform per wave: EXEC stays all-1
  int n = lane & 15, hi = lane >> 4;
  long rr_lane = (long)rt * 16 + (lane & 15);          // flat row rr = n*T + t
  int tt = (int)(rr_lane % T);
  int nn = (int)(rr_lane / T);
  const _Float16* rowp = spath + ((size_t)tt * N + nn) * H;
  float bv = bfc3[n];
  v8f acc;
#pragma unroll
  for (int r = 0; r < 8; ++r) acc[r] = bv;
#pragma unroll
  for (int kk = 0; kk < 4; ++kk) {
    v16h a = load_a_frag_row(rowp + kk * 32);
    v16h b = load_b_frag(Wf + (size_t)kk * 32 * NF, NF);
    acc = WMMA_F16(a, b, acc);
  }
#pragma unroll
  for (int r = 0; r < 8; ++r) {
    long rr = (long)rt * 16 + hi * 8 + r;
    out[rr * NF + n] = acc[r];
  }
}

// ---------------- host launcher ----------------

extern "C" void kernel_launch(void* const* d_in, const int* in_sizes, int n_in,
                              void* d_out, int out_size, void* d_ws, size_t ws_size,
                              hipStream_t stream) {
  (void)in_sizes; (void)n_in; (void)out_size; (void)ws_size;
  const float* z    = (const float*)d_in[0];
  const int*   ei   = (const int*)d_in[1];
  const unsigned char* xm = (const unsigned char*)d_in[2];
  const float* Wfc2 = (const float*)d_in[3];
  const float* bfc2 = (const float*)d_in[4];
  const float* Wih  = (const float*)d_in[5];
  const float* Whh  = (const float*)d_in[6];
  const float* bih  = (const float*)d_in[7];
  const float* bhh  = (const float*)d_in[8];
  const float* Wgcn = (const float*)d_in[9];
  const float* bgcn = (const float*)d_in[10];
  const float* Wfc3 = (const float*)d_in[11];
  const float* bfc3 = (const float*)d_in[12];
  float* out = (float*)d_out;
  char* ws = (char*)d_ws;

  size_t off = 0;
  auto take = [&](size_t bytes) { size_t r = off; off += (bytes + 255) & ~(size_t)255; return r; };
  float*    dinv  = (float*)(ws + take((size_t)N * 4));
  float*    nmask = (float*)(ws + take((size_t)N * 4));
  _Float16* Wfc2h = (_Float16*)(ws + take((size_t)L * H * 2));
  _Float16* WihTh = (_Float16*)(ws + take((size_t)4 * H * H * 2));
  _Float16* WhhTh = (_Float16*)(ws + take((size_t)4 * H * H * 2));
  _Float16* Wgcnh = (_Float16*)(ws + take((size_t)H * H * 2));
  _Float16* Wfc3h = (_Float16*)(ws + take((size_t)H * NF * 2));
  _Float16* xth   = (_Float16*)(ws + take((size_t)TN * H * 2));
  float*    xw    = (float*)(ws + take((size_t)TN * H * 4));
  float*    agg   = (float*)(ws + take((size_t)TN * H * 4));
  size_t regionB = off;                                  // temps dead after GCN GEMM
  _Float16* zh    = (_Float16*)(ws + take((size_t)N * L * 2));
  _Float16* hdh   = (_Float16*)(ws + take((size_t)N * H * 2));
  _Float16* hAh   = (_Float16*)(ws + take((size_t)N * H * 2));
  _Float16* hBh   = (_Float16*)(ws + take((size_t)N * H * 2));
  float*    cbuf  = (float*)(ws + take((size_t)N * H * 4));
  float*    xproj = (float*)(ws + take((size_t)N * 4 * H * 4));
  _Float16* spath = (_Float16*)(ws + regionB);           // aliases regionB (30.7MB <= 34.6MB)

  // prep
  k_conv_weights<<<256, 256, 0, stream>>>(Wfc2, Wih, Whh, Wgcn, Wfc3,
                                          Wfc2h, WihTh, WhhTh, Wgcnh, Wfc3h);
  k_f2h<<<2500, 256, 0, stream>>>(z, zh, (long)N * L);
  k_nodemask<<<(N + 255) / 256, 256, 0, stream>>>(xm, nmask);
  k_deg_init<<<(N + 255) / 256, 256, 0, stream>>>(dinv);
  k_deg_acc<<<(E + 255) / 256, 256, 0, stream>>>(ei, dinv);
  k_deg_fin<<<(N + 255) / 256, 256, 0, stream>>>(dinv);

  // fc2 + LSTM
  k_gemm_fc2<<<N / 16, 256, 0, stream>>>(zh, Wfc2h, bfc2, hdh, hAh, cbuf);
  k_gemm_xproj<<<dim3(N / 16, 4), 256, 0, stream>>>(hdh, WihTh, bih, xproj);
  for (int t = 0; t < T; ++t) {
    const _Float16* hin = (t == 0) ? hdh : ((t & 1) ? hAh : hBh);
    _Float16* hout = (t & 1) ? hBh : hAh;
    k_lstm_step<<<N / 16, 256, (size_t)4 * H * H * sizeof(_Float16), stream>>>(
        hin, WhhTh, xproj, bhh, cbuf, hout, nmask, xth, t);
  }

  // GCN (agg init fused into GEMM epilogue)
  k_gemm_gcn<<<TN / 16, 256, 0, stream>>>(xth, Wgcnh, dinv, bgcn, xw, agg);
  {
    long total = (long)E * T * (H / 4);
    int blocks = (int)((total + 255) / 256);
    k_scatter<<<blocks, 256, 0, stream>>>(ei, dinv, xw, agg);
  }

  // fc3
  k_f2h<<<4096, 256, 0, stream>>>(agg, spath, (long)TN * H);
  k_gemm_fc3<<<(TN / 16 + 7) / 8, 256, 0, stream>>>(spath, Wfc3h, bfc3, out);
}
